// MonteCarloNet_56994216018021
// MI455X (gfx1250) — matrix operand order repaired and checked by reading, not compiled, not run on hardware
//
#include <hip/hip_runtime.h>
#include <hip/hip_bf16.h>

typedef __attribute__((ext_vector_type(16))) __bf16        v16bf;
typedef __attribute__((ext_vector_type(8)))  float         v8f;
typedef __attribute__((ext_vector_type(4)))  unsigned int  v4u;
typedef __attribute__((ext_vector_type(2)))  short         v2s;
typedef __attribute__((ext_vector_type(2)))  float         v2f;

#define NMC_   1024
#define NPTS_  2048
#define NFEAT_ 64
#define RTOT_  (NMC_ * NPTS_)   // 2,097,152 flattened (m,n) rows

// ---------------- fast tanh (v_tanh_f32 on gfx1250 when available) ----------
__device__ __forceinline__ float tanh_fast(float x) {
#if defined(__has_builtin)
#if __has_builtin(__builtin_amdgcn_tanhf)
  return __builtin_amdgcn_tanhf(x);
#else
  float e = __expf(-2.0f * x);
  return (1.0f - e) / (1.0f + e);
#endif
#else
  float e = __expf(-2.0f * x);
  return (1.0f - e) / (1.0f + e);
#endif
}

// ---------------- counter-based hash RNG ------------------------------------
__device__ __forceinline__ unsigned int mix32(unsigned int x) {
  x ^= x >> 16; x *= 0x7feb352dU;
  x ^= x >> 15; x *= 0x846ca68bU;
  x ^= x >> 16;
  return x;
}

// Sign-replicate the two 16-bit halves: 0x0000 / 0xFFFF per half.
// short2 >> 15 lowers to v_pk_ashrrev_i16 (1 VALU op).
__device__ __forceinline__ unsigned int sign16_mask(unsigned int st) {
  union { unsigned int u; v2s s; } x;
  x.u = st;
  x.s = x.s >> 15;
  return x.u;
}

// Zero y unless bit `bit` of h is set: v_bfe_i32 + v_and_b32 (2 VALU).
__device__ __forceinline__ float mask_and(float y, unsigned int h, int bit) {
#if defined(__has_builtin)
#if __has_builtin(__builtin_amdgcn_sbfe)
  unsigned int am = (unsigned int)__builtin_amdgcn_sbfe((int)h, bit, 1);
#else
  unsigned int am = (unsigned int)(((int)(h << (31 - bit))) >> 31);
#endif
#else
  unsigned int am = (unsigned int)(((int)(h << (31 - bit))) >> 31);
#endif
  return __uint_as_float(am & __float_as_uint(y));
}

// ---------------- kernel 1: t1[n][f] = tanh(2 * input[n] * W1[f]) -----------
__global__ __launch_bounds__(256) void mc_t1_kernel(const float* __restrict__ inp,
                                                    const float* __restrict__ W1,
                                                    __bf16* __restrict__ t1) {
  int idx = blockIdx.x * 256 + threadIdx.x;   // exactly 2048*64 / 256 blocks
  int n = idx >> 6;
  int f = idx & 63;
  float h = inp[n] * W1[f];
  t1[idx] = (__bf16)tanh_fast(2.0f * h);
}

// A-fragment: 16x32 bf16, ISA layout (lanes0-15: K=hk8+0..7 in v0-3, 16+hk8+..
// in v4-7).  Per packed bf16 pair: one odd-constant multiply of the per-lane
// seed; the two 16-bit sign bits are the dropout bits (multiplicative hash),
// expanded with pk_ashrrev + and: 3 VALU per dword.
__device__ __forceinline__ v16bf load_a_frag(const __bf16* __restrict__ t1row,
                                             int ks0, unsigned int seed, int idx0) {
  union { v16bf v; unsigned int u[8]; v4u q[2]; } a;
  a.q[0] = *(const v4u*)(t1row + ks0);        // 8 contiguous bf16 (16B)
  a.q[1] = *(const v4u*)(t1row + ks0 + 16);   // 8 contiguous bf16 (16B)
#pragma unroll
  for (int d = 0; d < 8; ++d) {
    const unsigned int mult = 0x9E3779B9u + 0x3504F334u * (unsigned int)(idx0 + d);
    a.u[d] &= sign16_mask(seed * mult);       // dropout: zero bf16 halves
  }
  return a.v;
}

// ---------------- kernel 2: masked-GEMM(64x64) + mask2/tanh + dot(Wf) -------
__global__ __launch_bounds__(256) void mc_wmma_kernel(const __bf16* __restrict__ t1,
                                                      const float* __restrict__ W2,
                                                      const float* __restrict__ Wf,
                                                      float* __restrict__ out) {
  __shared__ __align__(128) __bf16 sW2[NFEAT_ * NFEAT_];   // 8 KB, B = 2*W2^T source
  __shared__ float sWf[NFEAT_];

  // Stage W2 into LDS with KEEP_SCALE=2 folded in: accs then hold 2*y directly.
  for (int i = threadIdx.x; i < NFEAT_ * NFEAT_; i += 256)
    sW2[i] = (__bf16)(2.0f * W2[i]);
  if (threadIdx.x < NFEAT_) sWf[threadIdx.x] = Wf[threadIdx.x];
  __syncthreads();

  const int lane = threadIdx.x & 31;
  const int wid  = threadIdx.x >> 5;           // 8 waves / block
  const int col  = lane & 15;
  const int halfbit = lane >> 4;               // 0 / 1 lane-half
  const int hk8  = halfbit * 8;                // A-frag K base per lane-half
  const int hk16 = halfbit * 16;               // unused; B-frag K base is hk8 too
  const int half8 = halfbit * 8;               // C/D row base per lane-half

  // Build the 8 B-fragments (4 N-blocks x 2 K-chunks) once; keep in VGPRs.
  v16bf Bf[4][2];
#pragma unroll
  for (int b = 0; b < 4; ++b)
#pragma unroll
    for (int c = 0; c < 2; ++c)
    {
      union { v16bf v; v4u q[2]; } bb;
      bb.q[0] = *(const v4u*)(sW2 + (16 * b + col) * NFEAT_ + 32 * c + hk8);
      bb.q[1] = *(const v4u*)(sW2 + (16 * b + col) * NFEAT_ + 32 * c + 16 + hk8);
      Bf[b][c] = bb.v;
    }

  // Broadcast Wf weights as {w,w} pairs once (feeds v_pk_fma_f32 later).
  v2f wf2[4];
#pragma unroll
  for (int b = 0; b < 4; ++b) {
    float w = sWf[16 * b + col];
    wf2[b].x = w; wf2[b].y = w;
  }

  const int stride = (int)gridDim.x * 128;     // 8 waves x 16 rows per block

  for (int base = (int)blockIdx.x * 128; base < RTOT_; base += stride) {
    const int r0 = base + wid * 16;            // this wave's 16-row tile
    const int r = r0 + col;                    // A-matrix M row for this lane
    const int n = r & (NPTS_ - 1);             // tile never crosses an MC sample
    const __bf16* t1row = t1 + n * NFEAT_;

    // Prefetch next tile's t1 row.
    int nb = base + stride;
    if (nb < RTOT_) {
      int nn = (nb + wid * 16 + col) & (NPTS_ - 1);
      __builtin_prefetch(t1 + nn * NFEAT_, 0, 1);
    }

    // One seed covers this lane's 32 layer-1 decisions (key unique per
    // (row, lane-half); every (row,feature) bit is consumed exactly once).
    const unsigned int seed1 =
        mix32(((unsigned int)r * 2u + (unsigned int)halfbit) * 0x01000193u + 0x243F6A88u);
    v16bf a0 = load_a_frag(t1row, 0 + hk8, seed1, 0);   // K 0..31
    v16bf a1 = load_a_frag(t1row, 32 + hk8, seed1, 8);  // K 32..63

    v8f accs[4] = {};
#pragma unroll
    for (int b = 0; b < 4; ++b) {
      accs[b] = __builtin_amdgcn_wmma_f32_16x16x32_bf16(
          false, a0, false, Bf[b][0], (short)0, accs[b], false, false);
      accs[b] = __builtin_amdgcn_wmma_f32_16x16x32_bf16(
          false, a1, false, Bf[b][1], (short)0, accs[b], false, false);
    }

    // Epilogue: mask2 -> tanh(2y already in acc) -> dot(Wf), accumulated in
    // float2 pairs so the 32 fma become 16 v_pk_fma_f32.  One hash gives all
    // 32 (row i, block b) decisions per lane at constant bits 4i+b.
    const unsigned int h2 = mix32((unsigned int)(r0 + half8) * 0x01000193u ^
                                  (unsigned int)col * 0x9E3779B1u ^ 0xB5297A4Du);
    v2f sp[4] = {};
#pragma unroll
    for (int b = 0; b < 4; ++b) {
#pragma unroll
      for (int p = 0; p < 4; ++p) {
        const int i0 = 2 * p, i1 = 2 * p + 1;
        v2f tt;
        tt.x = tanh_fast(mask_and(accs[b][i0], h2, 4 * i0 + b));
        tt.y = tanh_fast(mask_and(accs[b][i1], h2, 4 * i1 + b));
        sp[p] = tt * wf2[b] + sp[p];           // v_pk_fma_f32
      }
    }

    // Fold-as-you-go butterfly: 15 shuffles; stage 1 adds packed.
#pragma unroll
    for (int p = 0; p < 4; ++p) {
      v2f o;
      o.x = __shfl_xor(sp[p].x, 1, 32);
      o.y = __shfl_xor(sp[p].y, 1, 32);
      sp[p] += o;                              // v_pk_add_f32
    }
    const bool f0 = (lane & 1) != 0;
    float t[4];
#pragma unroll
    for (int j = 0; j < 4; ++j) t[j] = f0 ? sp[j].y : sp[j].x;
#pragma unroll
    for (int j = 0; j < 4; ++j) t[j] += __shfl_xor(t[j], 2, 32);
    const bool f1 = (lane & 2) != 0;
    float u0 = f1 ? t[1] : t[0];
    float u1 = f1 ? t[3] : t[2];
    u0 += __shfl_xor(u0, 4, 32);
    u1 += __shfl_xor(u1, 4, 32);
    const bool f2 = (lane & 4) != 0;
    float v = f2 ? u1 : u0;
    v += __shfl_xor(v, 8, 32);                 // lane holds sum for row (lane&7)

    if (col < 8) out[r0 + half8 + col] = v;    // 16 contiguous f32 stores / tile
  }
}

extern "C" void kernel_launch(void* const* d_in, const int* in_sizes, int n_in,
                              void* d_out, int out_size, void* d_ws, size_t ws_size,
                              hipStream_t stream) {
  (void)in_sizes; (void)n_in; (void)out_size; (void)ws_size;
  const float* input = (const float*)d_in[0];   // [2048,1]
  const float* W1    = (const float*)d_in[1];   // [64,1]
  const float* W2    = (const float*)d_in[2];   // [64,64]
  const float* Wf    = (const float*)d_in[3];   // [1,64]
  float* out = (float*)d_out;                   // [1024,2048,1]
  __bf16* t1 = (__bf16*)d_ws;                   // 2048*64 bf16 = 256 KB scratch

  // Kernel 1: precompute t1 = tanh(2 * input * W1^T)  (2048*64 elems)
  mc_t1_kernel<<<(NPTS_ * NFEAT_) / 256, 256, 0, stream>>>(input, W1, t1);

  // Kernel 2: persistent WMMA GEMM over all 2M MC rows (8 batches / block)
  mc_wmma_kernel<<<2048, 256, 0, stream>>>(t1, W2, Wf, out);
}
